// AHGCRU_27556510171436
// MI455X (gfx1250) — matrix operand order repaired and checked by reading, not compiled, not run on hardware
//
#include <hip/hip_runtime.h>

// ---------------------------------------------------------------------------
// AHGCRU for MI455X (gfx1250): bf16 WMMA GEMM pipeline, f32 accumulate.
// Each wave computes a 16x64 output strip (4 accumulators, shared A fragment),
// with explicit double-buffered fragment prefetch (software pipeline).
// ---------------------------------------------------------------------------

typedef __attribute__((ext_vector_type(16))) __bf16 v16bf;
typedef __attribute__((ext_vector_type(8)))  __bf16 v8bf;
typedef __attribute__((ext_vector_type(8)))  float  v8f;

constexpr int BATCH = 16;
constexpr int NNODE = 1000;
constexpr int FIN   = 32;
constexpr int TT    = 12;
constexpr int HID   = 64;
constexpr int OUTC  = 64;
constexpr int MEDGE = 500;
constexpr int EMB   = 16;
constexpr int NP    = 1024;   // padded node count (multiple of 16/32)
constexpr int MP    = 512;    // padded edge count
constexpr int CIN   = 96;     // F_IN + HID
constexpr int H2    = 128;    // 2*HID

__device__ __forceinline__ __bf16 f2bf(float f) {
    unsigned u = __builtin_bit_cast(unsigned, f);
    unsigned r = u + 0x7FFFu + ((u >> 16) & 1u);   // round-to-nearest-even
    unsigned short h = (unsigned short)(r >> 16);
    return __builtin_bit_cast(__bf16, h);
}

// ---------------------------------------------------------------------------
// Generic batched WMMA GEMM:  C[b] = rowScale ⊙ (A[b] @ B[b]) + colBias
//   A  : row-major (M x K), bf16, row stride lda, batch stride sAb
//   Bt : B stored transposed (Nn x K), bf16, row stride ldbt, batch stride sBb
//   C  : optional f32 out, element (m, n) at m*ldc + n*cstride (+ b*sCb)
//        rows with m >= rowMax skipped when rowMax > 0
//   CT : optional bf16 transposed out, element at n*ldct + m (+ b*sCTb)
// One wave (32 threads) computes a 16x64 strip: 4 column tiles sharing one
// A fragment per K-step. K is a compile-time constant (multiple of 32);
// fragments for step k+1 are prefetched before the WMMA chain of step k.
// Fragment k order: contiguous, half h holds k = 16h+s (same for A and B, wave32).
// ---------------------------------------------------------------------------
template <int K>
__global__ void wmma_gemm(const __bf16* __restrict__ A,  long long sAb, int lda,
                          const __bf16* __restrict__ Bt, long long sBb, int ldbt,
                          float* __restrict__ C, long long sCb, int ldc, int cstride, int rowMax,
                          __bf16* __restrict__ CT, long long sCTb, int ldct,
                          const float* __restrict__ rowScale,
                          const float* __restrict__ colBias)
{
    const int nt = blockIdx.x;      // 64-column strip index
    const int mt = blockIdx.y, b = blockIdx.z;
    const int lane = threadIdx.x & 31;
    const int lh = lane >> 4;       // half-wave select
    const int ll = lane & 15;
    const long long bstep = (long long)16 * ldbt;   // next 16-column tile
    const __bf16* Ap = A + (long long)b * sAb + (long long)(mt * 16 + ll) * lda + 16 * lh;
    const __bf16* Bp = Bt + (long long)b * sBb + (long long)(nt * 64 + ll) * ldbt + 16 * lh;

    v8f acc[4];
#pragma unroll
    for (int j = 0; j < 4; ++j)
#pragma unroll
        for (int i = 0; i < 8; ++i) acc[j][i] = 0.0f;

    // prologue: fragments for k0 = 0
    v8bf a0 = *(const v8bf*)(Ap);
    v8bf a1 = *(const v8bf*)(Ap + 8);
    v8bf b0[4], b1[4];
#pragma unroll
    for (int j = 0; j < 4; ++j) {
        b0[j] = *(const v8bf*)(Bp + (long long)j * bstep);
        b1[j] = *(const v8bf*)(Bp + (long long)j * bstep + 8);
    }

#pragma unroll (K <= 128 ? K / 32 : 2)
    for (int k0 = 0; k0 < K; k0 += 32) {
        // prefetch next K-step (clamped: last iteration re-reads current step)
        const int kn = (k0 + 32 < K) ? (k0 + 32) : k0;
        v8bf na0 = *(const v8bf*)(Ap + kn);
        v8bf na1 = *(const v8bf*)(Ap + kn + 8);
        v8bf nb0[4], nb1[4];
#pragma unroll
        for (int j = 0; j < 4; ++j) {
            nb0[j] = *(const v8bf*)(Bp + (long long)j * bstep + kn);
            nb1[j] = *(const v8bf*)(Bp + (long long)j * bstep + kn + 8);
        }

        v16bf av;
#pragma unroll
        for (int i = 0; i < 8; ++i) { av[i] = a0[i]; av[8 + i] = a1[i]; }
#pragma unroll
        for (int j = 0; j < 4; ++j) {
            v16bf bv;
#pragma unroll
            for (int i = 0; i < 8; ++i) { bv[i] = b0[j][i]; bv[8 + i] = b1[j][i]; }
            acc[j] = __builtin_amdgcn_wmma_f32_16x16x32_bf16(
                         false, av, false, bv, (short)0, acc[j], false, false);
        }
        a0 = na0; a1 = na1;
#pragma unroll
        for (int j = 0; j < 4; ++j) { b0[j] = nb0[j]; b1[j] = nb1[j]; }
    }

    const int mbase = mt * 16 + 8 * lh;      // first of 8 consecutive rows
#pragma unroll
    for (int j = 0; j < 4; ++j) {
        const int ncol = nt * 64 + j * 16 + ll;   // output column
        const float cb = colBias ? colBias[ncol] : 0.0f;
        float vals[8];
#pragma unroll
        for (int v = 0; v < 8; ++v) {
            float val = acc[j][v];
            if (rowScale) val *= rowScale[mbase + v];
            vals[v] = val + cb;
        }
        if (C) {
            float* Cr = C + (long long)b * sCb;
#pragma unroll
            for (int v = 0; v < 8; ++v) {
                int m = mbase + v;
                if (rowMax == 0 || m < rowMax)
                    Cr[(long long)m * ldc + (long long)ncol * cstride] = vals[v];
            }
        }
        if (CT) {
            v8bf o;
#pragma unroll
            for (int v = 0; v < 8; ++v) o[v] = f2bf(vals[v]);
            *(v8bf*)(CT + (long long)b * sCTb + (long long)ncol * ldct + mbase) = o;
        }
    }
}

// ---------------------------------------------------------------------------
// Setup kernels
// ---------------------------------------------------------------------------
__global__ void init_kernel(const float* __restrict__ nodevec,
                            const float* __restrict__ edgevec,
                            float* __restrict__ DE, float* __restrict__ EE,
                            float* __restrict__ h,  __bf16* __restrict__ hb16)
{
    long long i = (long long)blockIdx.x * blockDim.x + threadIdx.x;
    if (i < (long long)BATCH * NP * HID) { h[i] = 0.0f; hb16[i] = f2bf(0.0f); }
    if (i < (long long)NP * EMB) {
        int n = (int)(i / EMB), e = (int)(i % EMB);
        DE[i] = (n < NNODE) ? tanhf(2.0f * nodevec[n * EMB + e]) : 0.0f;
    }
    if (i < (long long)MP * EMB) {
        int m = (int)(i / EMB), e = (int)(i % EMB);
        EE[i] = (m < MEDGE) ? tanhf(2.0f * edgevec[m * EMB + e]) : 0.0f;
    }
}

__global__ void adj_kernel(const float* __restrict__ DE, const float* __restrict__ EE,
                           float* __restrict__ adjf, __bf16* __restrict__ adjb,
                           __bf16* __restrict__ Stb)
{
    long long i = (long long)blockIdx.x * blockDim.x + threadIdx.x;
    if (i >= (long long)NP * MP) return;
    int m = (int)(i % MP), n = (int)(i / MP);
    float a = 0.0f, s = 0.0f;
    if (n < NNODE && m < MEDGE) {
        float d = 0.0f;
#pragma unroll
        for (int e = 0; e < EMB; ++e) d += DE[n * EMB + e] * EE[m * EMB + e];
        float tv = tanhf(2.0f * d);
        if (tv > 0.0f) { a = tv; s = 1.0f; }
    }
    adjf[(long long)n * MP + m] = a;
    adjb[(long long)n * MP + m] = f2bf(a);
    Stb[(long long)m * NP + n]  = f2bf(s);
}

__global__ void deg_kernel(const float* __restrict__ adjf,
                           float* __restrict__ Binv, float* __restrict__ Dinv)
{
    int i = blockIdx.x * blockDim.x + threadIdx.x;
    if (i < MP) {
        float c = 0.0f;
        if (i < MEDGE)
            for (int n = 0; n < NNODE; ++n)
                c += (adjf[(long long)n * MP + i] > 0.0f) ? 1.0f : 0.0f;
        Binv[i] = (c > 0.0f) ? 1.0f / c : 0.0f;
    }
    if (i < NP) {
        float s = 0.0f;
        if (i < NNODE)
            for (int m = 0; m < MEDGE; ++m)
                s += adjf[(long long)i * MP + m];
        Dinv[i] = (s > 0.0f) ? 1.0f / s : 0.0f;
    }
}

// convert float (K x Nout) row-major -> bf16 transposed (Nout x K)
__global__ void tconv_kernel(const float* __restrict__ src, __bf16* __restrict__ dst,
                             int K, int Nout)
{
    int i = blockIdx.x * blockDim.x + threadIdx.x;
    if (i >= K * Nout) return;
    int k = i / Nout, n = i % Nout;
    dst[n * K + k] = f2bf(src[i]);
}

// ---------------------------------------------------------------------------
// Per-step kernels
// ---------------------------------------------------------------------------
__global__ void build_xs_kernel(const float* __restrict__ x, const float* __restrict__ h,
                                __bf16* __restrict__ xs, int t)
{
    long long i = (long long)blockIdx.x * blockDim.x + threadIdx.x;
    if (i >= (long long)BATCH * NP * CIN) return;
    int j = (int)(i % CIN);
    long long bn = i / CIN;
    int n = (int)(bn % NP), b = (int)(bn / NP);
    float val = 0.0f;
    if (n < NNODE) {
        val = (j < FIN)
            ? x[(((long long)b * NNODE + n) * FIN + j) * TT + t]
            : h[((long long)b * NP + n) * HID + (j - FIN)];
    }
    xs[i] = f2bf(val);
}

// gate: zr = sigmoid(LN(relu(res+o))); z stored; xc = concat(xt, r*h) built (bf16)
__global__ void gate_kernel(const float* __restrict__ resg, const float* __restrict__ og,
                            const float* __restrict__ h,    const float* __restrict__ x,
                            const float* __restrict__ gamma,const float* __restrict__ beta,
                            float* __restrict__ z, __bf16* __restrict__ xc, int t)
{
    int row = blockIdx.x;                 // 0 .. BATCH*NNODE-1
    int b = row / NNODE, n = row % NNODE;
    int lane = threadIdx.x;
    long long base = ((long long)b * NP + n) * H2;
    float v[4], s = 0.0f, s2 = 0.0f;
#pragma unroll
    for (int i = 0; i < 4; ++i) {
        int c = lane + 32 * i;
        float u = resg[base + c] + og[base + c];
        u = u > 0.0f ? u : 0.0f;
        v[i] = u; s += u; s2 += u * u;
    }
    for (int off = 16; off; off >>= 1) {
        s  += __shfl_xor(s,  off, 32);
        s2 += __shfl_xor(s2, off, 32);
    }
    float mu  = s * (1.0f / H2);
    float var = s2 * (1.0f / H2) - mu * mu;
    float inv = rsqrtf(var + 1e-5f);
    float zr[4];
#pragma unroll
    for (int i = 0; i < 4; ++i) {
        int c = lane + 32 * i;
        float y = (v[i] - mu) * inv * gamma[c] + beta[c];
        zr[i] = 1.0f / (1.0f + expf(-y));
    }
    long long hbase = ((long long)b * NP + n) * HID;
    z[hbase + lane]      = zr[0];         // channel lane
    z[hbase + lane + 32] = zr[1];         // channel lane+32
    long long xcb = ((long long)b * NP + n) * CIN;
    xc[xcb + lane] = f2bf(x[(((long long)b * NNODE + n) * FIN + lane) * TT + t]);
    xc[xcb + FIN + lane]      = f2bf(zr[2] * h[hbase + lane]);        // r[lane]
    xc[xcb + FIN + lane + 32] = f2bf(zr[3] * h[hbase + lane + 32]);   // r[lane+32]
}

// candidate + state update: hc = tanh(LN(relu(res+o))); h = (1-z)h + z*hc
__global__ void update_kernel(const float* __restrict__ resc, const float* __restrict__ oc,
                              const float* __restrict__ z,
                              const float* __restrict__ gamma, const float* __restrict__ beta,
                              float* __restrict__ h, __bf16* __restrict__ hb16)
{
    int row = blockIdx.x;
    int b = row / NNODE, n = row % NNODE;
    int lane = threadIdx.x;
    long long base = ((long long)b * NP + n) * HID;
    float v[2], s = 0.0f, s2 = 0.0f;
#pragma unroll
    for (int i = 0; i < 2; ++i) {
        int c = lane + 32 * i;
        float u = resc[base + c] + oc[base + c];
        u = u > 0.0f ? u : 0.0f;
        v[i] = u; s += u; s2 += u * u;
    }
    for (int off = 16; off; off >>= 1) {
        s  += __shfl_xor(s,  off, 32);
        s2 += __shfl_xor(s2, off, 32);
    }
    float mu  = s * (1.0f / HID);
    float var = s2 * (1.0f / HID) - mu * mu;
    float inv = rsqrtf(var + 1e-5f);
#pragma unroll
    for (int i = 0; i < 2; ++i) {
        int c = lane + 32 * i;
        float y  = (v[i] - mu) * inv * gamma[c] + beta[c];
        float hc = tanhf(y);
        float zz = z[base + c];
        float hn = (1.0f - zz) * h[base + c] + zz * hc;
        h[base + c]    = hn;
        hb16[base + c] = f2bf(hn);
    }
}

// ---------------------------------------------------------------------------
// Launch
// ---------------------------------------------------------------------------
extern "C" void kernel_launch(void* const* d_in, const int* in_sizes, int n_in,
                              void* d_out, int out_size, void* d_ws, size_t ws_size,
                              hipStream_t stream)
{
    const float* x        = (const float*)d_in[0];
    const float* nodevec  = (const float*)d_in[1];
    const float* edgevec  = (const float*)d_in[2];
    const float* w_lin_g  = (const float*)d_in[3];
    const float* b_lin_g  = (const float*)d_in[4];
    const float* ln_g_w   = (const float*)d_in[5];
    const float* ln_g_b   = (const float*)d_in[6];
    const float* w_res_g  = (const float*)d_in[7];
    const float* b_res_g  = (const float*)d_in[8];
    const float* w_lin_c  = (const float*)d_in[9];
    const float* b_lin_c  = (const float*)d_in[10];
    const float* ln_c_w   = (const float*)d_in[11];
    const float* ln_c_b   = (const float*)d_in[12];
    const float* w_res_c  = (const float*)d_in[13];
    const float* b_res_c  = (const float*)d_in[14];
    const float* conv_w   = (const float*)d_in[15];
    const float* conv_b   = (const float*)d_in[16];
    float* out = (float*)d_out;

    // bump allocator over workspace (256B aligned)
    char* wsp = (char*)d_ws;
    auto alloc = [&](size_t bytes) -> void* {
        void* p = (void*)wsp;
        wsp += (bytes + 255) & ~(size_t)255;
        return p;
    };
    float*  DE    = (float*) alloc((size_t)NP * EMB * 4);
    float*  EE    = (float*) alloc((size_t)MP * EMB * 4);
    float*  Binv  = (float*) alloc((size_t)MP * 4);
    float*  Dinv  = (float*) alloc((size_t)NP * 4);
    float*  adjf  = (float*) alloc((size_t)NP * MP * 4);
    __bf16* adjb  = (__bf16*)alloc((size_t)NP * MP * 2);
    __bf16* Stb   = (__bf16*)alloc((size_t)MP * NP * 2);
    __bf16* WgT   = (__bf16*)alloc((size_t)H2  * CIN * 2);
    __bf16* WrgT  = (__bf16*)alloc((size_t)H2  * CIN * 2);
    __bf16* WcT   = (__bf16*)alloc((size_t)HID * CIN * 2);
    __bf16* WrcT  = (__bf16*)alloc((size_t)HID * CIN * 2);
    __bf16* cwT   = (__bf16*)alloc((size_t)HID * OUTC * 2);
    __bf16* xs    = (__bf16*)alloc((size_t)BATCH * NP * CIN * 2);
    __bf16* xc    = (__bf16*)alloc((size_t)BATCH * NP * CIN * 2);
    __bf16* hlTg  = (__bf16*)alloc((size_t)H2  * BATCH * NP * 2);
    __bf16* hlTc  = (__bf16*)alloc((size_t)HID * BATCH * NP * 2);
    float*  resg  = (float*) alloc((size_t)BATCH * NP * H2 * 4);
    float*  resc  = (float*) alloc((size_t)BATCH * NP * HID * 4);
    __bf16* ygT   = (__bf16*)alloc((size_t)H2  * BATCH * MP * 2);
    __bf16* ycT   = (__bf16*)alloc((size_t)HID * BATCH * MP * 2);
    float*  og    = (float*) alloc((size_t)BATCH * NP * H2 * 4);
    float*  oc    = (float*) alloc((size_t)BATCH * NP * HID * 4);
    float*  zbuf  = (float*) alloc((size_t)BATCH * NP * HID * 4);
    float*  h     = (float*) alloc((size_t)BATCH * NP * HID * 4);
    __bf16* hb16  = (__bf16*)alloc((size_t)BATCH * NP * HID * 2);

    auto cdiv = [](long long a, long long b) { return (unsigned)((a + b - 1) / b); };

    // ---- setup --------------------------------------------------------------
    init_kernel<<<cdiv((long long)BATCH * NP * HID, 256), 256, 0, stream>>>(
        nodevec, edgevec, DE, EE, h, hb16);
    adj_kernel<<<cdiv((long long)NP * MP, 256), 256, 0, stream>>>(DE, EE, adjf, adjb, Stb);
    deg_kernel<<<cdiv(NP, 256), 256, 0, stream>>>(adjf, Binv, Dinv);
    tconv_kernel<<<cdiv(CIN * H2,  256), 256, 0, stream>>>(w_lin_g, WgT,  CIN, H2);
    tconv_kernel<<<cdiv(CIN * H2,  256), 256, 0, stream>>>(w_res_g, WrgT, CIN, H2);
    tconv_kernel<<<cdiv(CIN * HID, 256), 256, 0, stream>>>(w_lin_c, WcT,  CIN, HID);
    tconv_kernel<<<cdiv(CIN * HID, 256), 256, 0, stream>>>(w_res_c, WrcT, CIN, HID);
    tconv_kernel<<<cdiv(HID * OUTC,256), 256, 0, stream>>>(conv_w,  cwT,  HID, OUTC);

    const int MROWS = BATCH * NP;        // 16384 flat rows
    // ---- recurrence ---------------------------------------------------------
    for (int t = 0; t < TT; ++t) {
        build_xs_kernel<<<cdiv((long long)MROWS * CIN, 256), 256, 0, stream>>>(x, h, xs, t);

        // hl_g^T = (xs @ Wg)^T  (bf16 transposed only)
        wmma_gemm<CIN><<<dim3(H2/64, MROWS/16, 1), 32, 0, stream>>>(
            xs, 0, CIN,  WgT, 0, CIN,
            nullptr, 0, 0, 1, 0,  hlTg, 0, MROWS,
            nullptr, nullptr);
        // res_g = xs @ Wrg + b_res_g  (f32)
        wmma_gemm<CIN><<<dim3(H2/64, MROWS/16, 1), 32, 0, stream>>>(
            xs, 0, CIN,  WrgT, 0, CIN,
            resg, 0, H2, 1, 0,  nullptr, 0, 0,
            nullptr, b_res_g);
        // y_g^T = Binv ⊙ (S^T @ hl_g), batched over B
        wmma_gemm<NP><<<dim3(H2/64, MP/16, BATCH), 32, 0, stream>>>(
            Stb, 0, NP,  hlTg, NP, MROWS,
            nullptr, 0, 0, 1, 0,  ygT, MP, (long long)BATCH * MP,
            Binv, nullptr);
        // o_g = Dinv ⊙ (adj @ y_g) + b_lin_g
        wmma_gemm<MP><<<dim3(H2/64, NP/16, BATCH), 32, 0, stream>>>(
            adjb, 0, MP,  ygT, MP, (long long)BATCH * MP,
            og, (long long)NP * H2, H2, 1, 0,  nullptr, 0, 0,
            Dinv, b_lin_g);

        gate_kernel<<<BATCH * NNODE, 32, 0, stream>>>(
            resg, og, h, x, ln_g_w, ln_g_b, zbuf, xc, t);

        // hl_c^T = (xc @ Wc)^T
        wmma_gemm<CIN><<<dim3(HID/64, MROWS/16, 1), 32, 0, stream>>>(
            xc, 0, CIN,  WcT, 0, CIN,
            nullptr, 0, 0, 1, 0,  hlTc, 0, MROWS,
            nullptr, nullptr);
        // res_c = xc @ Wrc + b_res_c
        wmma_gemm<CIN><<<dim3(HID/64, MROWS/16, 1), 32, 0, stream>>>(
            xc, 0, CIN,  WrcT, 0, CIN,
            resc, 0, HID, 1, 0,  nullptr, 0, 0,
            nullptr, b_res_c);
        // y_c^T = Binv ⊙ (S^T @ hl_c)
        wmma_gemm<NP><<<dim3(HID/64, MP/16, BATCH), 32, 0, stream>>>(
            Stb, 0, NP,  hlTc, NP, MROWS,
            nullptr, 0, 0, 1, 0,  ycT, MP, (long long)BATCH * MP,
            Binv, nullptr);
        // o_c = Dinv ⊙ (adj @ y_c) + b_lin_c
        wmma_gemm<MP><<<dim3(HID/64, NP/16, BATCH), 32, 0, stream>>>(
            adjb, 0, MP,  ycT, MP, (long long)BATCH * MP,
            oc, (long long)NP * HID, HID, 1, 0,  nullptr, 0, 0,
            Dinv, b_lin_c);

        update_kernel<<<BATCH * NNODE, 32, 0, stream>>>(
            resc, oc, zbuf, ln_c_w, ln_c_b, h, hb16);

        // out[b,n,:,t] = h @ conv_w + conv_b  (strided write into d_out)
        wmma_gemm<HID><<<dim3(OUTC/64, NP/16, BATCH), 32, 0, stream>>>(
            hb16, (long long)NP * HID, HID,  cwT, 0, HID,
            out + t, (long long)NNODE * OUTC * TT, OUTC * TT, TT, NNODE,
            nullptr, 0, 0,
            nullptr, conv_b);
    }
}
